// TransformerEncoder_86672440033609
// MI455X (gfx1250) — matrix operand order repaired and checked by reading, not compiled, not run on hardware
//
#include <hip/hip_runtime.h>
#include <hip/hip_bf16.h>
#include <math.h>

// ---------------- problem constants ----------------
#define BB   1024
#define SS   30
#define DD   512
#define HH   8
#define DKK  64
#define LLAYERS 4
#define MROWS (BB*SS)   // 30720
#define KPAD 320        // E=300 padded to 320 (multiple of 64)

// ---------------- WMMA types ----------------
typedef __bf16 v16bf __attribute__((ext_vector_type(16)));
typedef float  v8f   __attribute__((ext_vector_type(8)));

union FragAB { uint4 q[2]; v16bf v; };

__device__ __forceinline__ unsigned short f2b(float f) {
    unsigned int u = __float_as_uint(f);
    u += 0x7FFFu + ((u >> 16) & 1u);          // round-to-nearest-even
    return (unsigned short)(u >> 16);
}
__device__ __forceinline__ float b2f(unsigned short h) {
    return __uint_as_float(((unsigned int)h) << 16);
}

// ======================================================================
// Generic bf16 GEMM:  C[M,N] = A[M,K] * W[N,K]^T   (nn.Linear convention)
// 128x128 block tile, BK=64, 8 waves, wave tile 32x64 (2x4 WMMA tiles)
// Epilogue: optional bias, optional (scale*sqrt(D) + positional encoding),
//           writes f32 and/or bf16.
// ======================================================================
#define BM 128
#define BN 128
#define BK 64
#define LDK 72   // padded LDS row stride (keeps b128 alignment, avoids conflicts)

__global__ __launch_bounds__(256)
void gemm_bf16_kernel(const unsigned short* __restrict__ A,
                      const unsigned short* __restrict__ W,
                      float* __restrict__ Cf,
                      unsigned short* __restrict__ Cb,
                      const float* __restrict__ bias,
                      int M, int N, int K, int posenc)
{
    __shared__ unsigned short As[BM][LDK];
    __shared__ unsigned short Bs[BN][LDK];

    const int tid  = threadIdx.x;
    const int lane = tid & 31;
    const int w    = tid >> 5;
    const int l15  = lane & 15;
    const int half = lane >> 4;
    const int m0 = blockIdx.x * BM;
    const int n0 = blockIdx.y * BN;
    const int wm = w >> 1;   // 0..3  -> 32 rows each
    const int wn = w & 1;    // 0..1  -> 64 cols each

    v8f acc[2][4];
#pragma unroll
    for (int i = 0; i < 2; i++)
#pragma unroll
        for (int j = 0; j < 4; j++)
#pragma unroll
            for (int e = 0; e < 8; e++) acc[i][j][e] = 0.f;

    const int lr = tid >> 1;          // loader row 0..127
    const int lc = (tid & 1) * 32;    // loader col 0 / 32 (elements)

    for (int k0 = 0; k0 < K; k0 += BK) {
        const uint4* gA = (const uint4*)(A + (size_t)(m0 + lr) * K + k0 + lc);
        const uint4* gB = (const uint4*)(W + (size_t)(n0 + lr) * K + k0 + lc);
        uint4 a0 = gA[0], a1 = gA[1], a2 = gA[2], a3 = gA[3];
        uint4 b0 = gB[0], b1 = gB[1], b2 = gB[2], b3 = gB[3];
        if (k0 + BK < K) {   // hint: global_prefetch for next k-tile
            __builtin_prefetch((const void*)(A + (size_t)(m0 + lr) * K + k0 + BK + lc), 0, 1);
            __builtin_prefetch((const void*)(W + (size_t)(n0 + lr) * K + k0 + BK + lc), 0, 1);
        }
        *(uint4*)&As[lr][lc +  0] = a0;
        *(uint4*)&As[lr][lc +  8] = a1;
        *(uint4*)&As[lr][lc + 16] = a2;
        *(uint4*)&As[lr][lc + 24] = a3;
        *(uint4*)&Bs[lr][lc +  0] = b0;
        *(uint4*)&Bs[lr][lc +  8] = b1;
        *(uint4*)&Bs[lr][lc + 16] = b2;
        *(uint4*)&Bs[lr][lc + 24] = b3;
        __syncthreads();

#pragma unroll
        for (int ks = 0; ks < BK; ks += 32) {
            FragAB af[2], bf[4];
#pragma unroll
            for (int mi = 0; mi < 2; mi++) {
                int r = wm * 32 + mi * 16 + l15;           // A row (M)
                af[mi].q[0] = *(const uint4*)&As[r][ks + half * 8];
                af[mi].q[1] = *(const uint4*)&As[r][ks + 16 + half * 8];
            }
#pragma unroll
            for (int ni = 0; ni < 4; ni++) {
                int r = wn * 64 + ni * 16 + l15;           // B row (N)
                bf[ni].q[0] = *(const uint4*)&Bs[r][ks + half * 8];
                bf[ni].q[1] = *(const uint4*)&Bs[r][ks + 16 + half * 8];
            }
#pragma unroll
            for (int mi = 0; mi < 2; mi++)
#pragma unroll
                for (int ni = 0; ni < 4; ni++)
                    acc[mi][ni] = __builtin_amdgcn_wmma_f32_16x16x32_bf16(
                        false, af[mi].v, false, bf[ni].v,
                        (short)0, acc[mi][ni], false, false);
        }
        __syncthreads();
    }

    const float sqrtD = 22.62741699796952f;              // sqrt(512)
    const float negln = -9.210340371976184f / 512.0f;    // -ln(10000)/D
#pragma unroll
    for (int mi = 0; mi < 2; mi++)
#pragma unroll
        for (int ni = 0; ni < 4; ni++) {
            int n = n0 + wn * 64 + ni * 16 + l15;
            float bv = bias ? bias[n] : 0.f;
#pragma unroll
            for (int r = 0; r < 8; r++) {
                int m = m0 + wm * 32 + mi * 16 + half * 8 + r;
                float v = acc[mi][ni][r] + bv;
                if (posenc) {
                    int s  = m % SS;
                    int de = n & ~1;
                    float div = __expf((float)de * negln);
                    float ang = (float)s * div;
                    float pe  = (n & 1) ? __cosf(ang) : __sinf(ang);
                    v = v * sqrtD + pe;
                }
                size_t off = (size_t)m * N + n;
                if (Cf) Cf[off] = v;
                if (Cb) Cb[off] = f2b(v);
            }
        }
}

// ======================================================================
// Per-head QKV GEMM: for head h, C[:, h*64+d] = sum_k H[:, h*64+k]*Wx[d,k]
// grid = (M/128, 8 heads, 3 {q,k,v}); wave tile 32x32 (2x2 WMMA tiles)
// ======================================================================
__global__ __launch_bounds__(256)
void gemm_qkv_kernel(const unsigned short* __restrict__ Hb,
                     const unsigned short* __restrict__ Wq,
                     const unsigned short* __restrict__ Wk,
                     const unsigned short* __restrict__ Wv,
                     unsigned short* __restrict__ Qb,
                     unsigned short* __restrict__ Kb,
                     unsigned short* __restrict__ Vb)
{
    __shared__ unsigned short As[BM][LDK];
    __shared__ unsigned short Bs[DKK][LDK];

    const unsigned short* W = (blockIdx.z == 0) ? Wq : ((blockIdx.z == 1) ? Wk : Wv);
    unsigned short*       C = (blockIdx.z == 0) ? Qb : ((blockIdx.z == 1) ? Kb : Vb);
    const int head = blockIdx.y;
    const int m0   = blockIdx.x * BM;

    const int tid  = threadIdx.x;
    const int lane = tid & 31;
    const int w    = tid >> 5;
    const int l15  = lane & 15;
    const int half = lane >> 4;
    const int wm = w >> 1;   // 0..3
    const int wn = w & 1;    // 0..1

    // load A: 128 rows x 64 cols from column band head*64
    {
        int lr = tid >> 1, lc = (tid & 1) * 32;
        const uint4* gA = (const uint4*)(Hb + (size_t)(m0 + lr) * DD + head * DKK + lc);
        uint4 a0 = gA[0], a1 = gA[1], a2 = gA[2], a3 = gA[3];
        *(uint4*)&As[lr][lc +  0] = a0;
        *(uint4*)&As[lr][lc +  8] = a1;
        *(uint4*)&As[lr][lc + 16] = a2;
        *(uint4*)&As[lr][lc + 24] = a3;
    }
    // load B: 64x64 weight
    {
        int br = tid >> 2, bc = (tid & 3) * 16;
        const uint4* gB = (const uint4*)(W + (size_t)br * DKK + bc);
        uint4 b0 = gB[0], b1 = gB[1];
        *(uint4*)&Bs[br][bc + 0] = b0;
        *(uint4*)&Bs[br][bc + 8] = b1;
    }
    __syncthreads();

    v8f acc[2][2];
#pragma unroll
    for (int i = 0; i < 2; i++)
#pragma unroll
        for (int j = 0; j < 2; j++)
#pragma unroll
            for (int e = 0; e < 8; e++) acc[i][j][e] = 0.f;

#pragma unroll
    for (int ks = 0; ks < DKK; ks += 32) {
        FragAB af[2], bf[2];
#pragma unroll
        for (int mi = 0; mi < 2; mi++) {
            int r = wm * 32 + mi * 16 + l15;
            af[mi].q[0] = *(const uint4*)&As[r][ks + half * 8];
            af[mi].q[1] = *(const uint4*)&As[r][ks + 16 + half * 8];
        }
#pragma unroll
        for (int ni = 0; ni < 2; ni++) {
            int r = wn * 32 + ni * 16 + l15;
            bf[ni].q[0] = *(const uint4*)&Bs[r][ks + half * 8];
            bf[ni].q[1] = *(const uint4*)&Bs[r][ks + 16 + half * 8];
        }
#pragma unroll
        for (int mi = 0; mi < 2; mi++)
#pragma unroll
            for (int ni = 0; ni < 2; ni++)
                acc[mi][ni] = __builtin_amdgcn_wmma_f32_16x16x32_bf16(
                    false, af[mi].v, false, bf[ni].v,
                    (short)0, acc[mi][ni], false, false);
    }

#pragma unroll
    for (int mi = 0; mi < 2; mi++)
#pragma unroll
        for (int ni = 0; ni < 2; ni++) {
            int nl = wn * 32 + ni * 16 + l15;
#pragma unroll
            for (int r = 0; r < 8; r++) {
                int m = m0 + wm * 32 + mi * 16 + half * 8 + r;
                C[(size_t)m * DD + head * DKK + nl] = f2b(acc[mi][ni][r]);
            }
        }
}

// ======================================================================
// Attention: one wave per (b,h).  S=30, d_k=64.  K/V staged in LDS (f32).
// block = 64 threads = 2 waves = 2 (b,h) pairs.
// ======================================================================
__global__ __launch_bounds__(64)
void attn_kernel(const unsigned short* __restrict__ Qb,
                 const unsigned short* __restrict__ Kb,
                 const unsigned short* __restrict__ Vb,
                 const int* __restrict__ mask,
                 unsigned short* __restrict__ Ob)
{
    __shared__ float kf[2][SS][DKK];
    __shared__ float vf[2][SS][DKK];
    __shared__ float madd[2][32];

    const int w    = threadIdx.x >> 5;
    const int lane = threadIdx.x & 31;
    const int p = blockIdx.x * 2 + w;
    const int b = p >> 3;
    const int h = p & 7;
    const size_t base = (size_t)b * SS * DD + (size_t)h * DKK;

#pragma unroll
    for (int j = 0; j < SS; j++) {
        kf[w][j][lane]      = b2f(Kb[base + (size_t)j * DD + lane]);
        kf[w][j][lane + 32] = b2f(Kb[base + (size_t)j * DD + lane + 32]);
        vf[w][j][lane]      = b2f(Vb[base + (size_t)j * DD + lane]);
        vf[w][j][lane + 32] = b2f(Vb[base + (size_t)j * DD + lane + 32]);
    }
    if (lane < SS) madd[w][lane] = (mask[b * SS + lane] == 0) ? -1e9f : 0.f;
    __syncthreads();

    if (lane < SS) {
        float qr[DKK];
#pragma unroll
        for (int d = 0; d < DKK; d++) qr[d] = b2f(Qb[base + (size_t)lane * DD + d]);

        float sc[SS];
        float mx = -3.4e38f;
#pragma unroll
        for (int j = 0; j < SS; j++) {
            float s = 0.f;
#pragma unroll
            for (int d = 0; d < DKK; d++) s += qr[d] * kf[w][j][d];
            s = s * 0.125f + madd[w][j];        // 1/sqrt(64)
            sc[j] = s;
            mx = fmaxf(mx, s);
        }
        float sum = 0.f;
#pragma unroll
        for (int j = 0; j < SS; j++) { sc[j] = __expf(sc[j] - mx); sum += sc[j]; }
        float inv = 1.f / sum;

        float o[DKK];
#pragma unroll
        for (int d = 0; d < DKK; d++) o[d] = 0.f;
#pragma unroll
        for (int j = 0; j < SS; j++) {
            float pj = sc[j] * inv;
#pragma unroll
            for (int d = 0; d < DKK; d++) o[d] += pj * vf[w][j][d];
        }
#pragma unroll
        for (int d = 0; d < DKK; d++)
            Ob[base + (size_t)lane * DD + d] = f2b(o[d]);
    }
}

// ======================================================================
// Fused residual + LayerNorm: out = LN(X + R)*g + beta.  One wave per row.
// ======================================================================
__global__ __launch_bounds__(256)
void ln_kernel(const float* __restrict__ X, const float* __restrict__ R,
               const float* __restrict__ g, const float* __restrict__ beta,
               float* __restrict__ Of, unsigned short* __restrict__ Ob)
{
    const int w    = threadIdx.x >> 5;
    const int lane = threadIdx.x & 31;
    const int row  = blockIdx.x * 8 + w;
    const float* xr = X + (size_t)row * DD;
    const float* rr = R + (size_t)row * DD;

    float vals[16];
    float s = 0.f, sq = 0.f;
#pragma unroll
    for (int i = 0; i < 16; i++) {
        float v = xr[lane + i * 32] + rr[lane + i * 32];
        vals[i] = v; s += v; sq += v * v;
    }
#pragma unroll
    for (int off = 16; off >= 1; off >>= 1) {
        s  += __shfl_xor(s,  off, 32);
        sq += __shfl_xor(sq, off, 32);
    }
    float mu  = s  * (1.f / (float)DD);
    float var = sq * (1.f / (float)DD) - mu * mu;
    float rs  = rsqrtf(var + 1e-5f);
#pragma unroll
    for (int i = 0; i < 16; i++) {
        int d = lane + i * 32;
        float v = (vals[i] - mu) * rs * g[d] + beta[d];
        Of[(size_t)row * DD + d] = v;
        if (Ob) Ob[(size_t)row * DD + d] = f2b(v);
    }
}

// ======================================================================
// Conversion / staging kernels
// ======================================================================
__global__ void cvt_bf16_kernel(const float* __restrict__ src,
                                unsigned short* __restrict__ dst, size_t n)
{
    size_t i = (size_t)blockIdx.x * blockDim.x + threadIdx.x;
    if (i < n) dst[i] = f2b(src[i]);
}

__global__ void pad_win_kernel(const float* __restrict__ W_in,
                               unsigned short* __restrict__ dst)   // [512][320]
{
    size_t i = (size_t)blockIdx.x * blockDim.x + threadIdx.x;
    if (i < (size_t)DD * KPAD) {
        int r = (int)(i / KPAD), c = (int)(i % KPAD);
        dst[i] = (c < 300) ? f2b(W_in[(size_t)r * 300 + c]) : (unsigned short)0;
    }
}

__global__ void gather_emb_kernel(const int* __restrict__ x,
                                  const float* __restrict__ emb,
                                  unsigned short* __restrict__ dst) // [30720][320]
{
    size_t i = (size_t)blockIdx.x * blockDim.x + threadIdx.x;
    if (i < (size_t)MROWS * KPAD) {
        int m = (int)(i / KPAD), c = (int)(i % KPAD);
        dst[i] = (c < 300) ? f2b(emb[(size_t)x[m] * 300 + c]) : (unsigned short)0;
    }
}

// ======================================================================
extern "C" void kernel_launch(void* const* d_in, const int* in_sizes, int n_in,
                              void* d_out, int out_size, void* d_ws, size_t ws_size,
                              hipStream_t stream)
{
    const int*   x     = (const int*)d_in[0];
    const int*   mask  = (const int*)d_in[1];
    const float* emb   = (const float*)d_in[2];
    const float* W_in  = (const float*)d_in[3];
    const float* b_in  = (const float*)d_in[4];
    const float* Wq    = (const float*)d_in[5];
    const float* Wk    = (const float*)d_in[6];
    const float* Wv    = (const float*)d_in[7];
    const float* Wo    = (const float*)d_in[8];
    const float* Wfc   = (const float*)d_in[9];
    const float* bfc   = (const float*)d_in[10];
    const float* g1    = (const float*)d_in[11];
    const float* beta1 = (const float*)d_in[12];
    const float* g2    = (const float*)d_in[13];
    const float* beta2 = (const float*)d_in[14];
    float* hF = (float*)d_out;   // h lives in d_out (f32) — final result ends here

    char* p = (char*)d_ws;
    auto take = [&](size_t bytes) -> char* {
        char* r = p; p += (bytes + 255) & ~(size_t)255; return r;
    };
    unsigned short* Aemb = (unsigned short*)take((size_t)MROWS * KPAD * 2);
    unsigned short* WinB = (unsigned short*)take((size_t)DD * KPAD * 2);
    unsigned short* WqB  = (unsigned short*)take((size_t)LLAYERS * DKK * DKK * 2);
    unsigned short* WkB  = (unsigned short*)take((size_t)LLAYERS * DKK * DKK * 2);
    unsigned short* WvB  = (unsigned short*)take((size_t)LLAYERS * DKK * DKK * 2);
    unsigned short* WoB  = (unsigned short*)take((size_t)LLAYERS * DD * DD * 2);
    unsigned short* WfcB = (unsigned short*)take((size_t)LLAYERS * DD * DD * 2);
    unsigned short* hB   = (unsigned short*)take((size_t)MROWS * DD * 2);
    unsigned short* n1B  = (unsigned short*)take((size_t)MROWS * DD * 2);
    unsigned short* oB   = (unsigned short*)take((size_t)MROWS * DD * 2);
    unsigned short* qB   = (unsigned short*)take((size_t)MROWS * DD * 2);
    unsigned short* kB   = (unsigned short*)take((size_t)MROWS * DD * 2);
    unsigned short* vB   = (unsigned short*)take((size_t)MROWS * DD * 2);
    float* tF  = (float*)take((size_t)MROWS * DD * 4);
    float* n1F = (float*)take((size_t)MROWS * DD * 4);

    // ---- weight conversion ----
    {
        size_t n = (size_t)LLAYERS * DKK * DKK;
        cvt_bf16_kernel<<<(unsigned)((n + 255) / 256), 256, 0, stream>>>(Wq, WqB, n);
        cvt_bf16_kernel<<<(unsigned)((n + 255) / 256), 256, 0, stream>>>(Wk, WkB, n);
        cvt_bf16_kernel<<<(unsigned)((n + 255) / 256), 256, 0, stream>>>(Wv, WvB, n);
        size_t nw = (size_t)LLAYERS * DD * DD;
        cvt_bf16_kernel<<<(unsigned)((nw + 255) / 256), 256, 0, stream>>>(Wo,  WoB,  nw);
        cvt_bf16_kernel<<<(unsigned)((nw + 255) / 256), 256, 0, stream>>>(Wfc, WfcB, nw);
        size_t np = (size_t)DD * KPAD;
        pad_win_kernel<<<(unsigned)((np + 255) / 256), 256, 0, stream>>>(W_in, WinB);
        size_t ng = (size_t)MROWS * KPAD;
        gather_emb_kernel<<<(unsigned)((ng + 255) / 256), 256, 0, stream>>>(x, emb, Aemb);
    }

    // ---- input projection + scale*sqrt(D) + positional encoding ----
    gemm_bf16_kernel<<<dim3(MROWS / BM, DD / BN), 256, 0, stream>>>(
        Aemb, WinB, hF, hB, b_in, MROWS, DD, KPAD, 1);

    // ---- transformer blocks ----
    for (int l = 0; l < LLAYERS; ++l) {
        gemm_qkv_kernel<<<dim3(MROWS / BM, HH, 3), 256, 0, stream>>>(
            hB, WqB + (size_t)l * DKK * DKK, WkB + (size_t)l * DKK * DKK,
            WvB + (size_t)l * DKK * DKK, qB, kB, vB);

        attn_kernel<<<(BB * HH) / 2, 64, 0, stream>>>(qB, kB, vB, mask, oB);

        gemm_bf16_kernel<<<dim3(MROWS / BM, DD / BN), 256, 0, stream>>>(
            oB, WoB + (size_t)l * DD * DD, tF, nullptr, nullptr, MROWS, DD, DD, 0);

        ln_kernel<<<MROWS / 8, 256, 0, stream>>>(
            tF, hF, g1 + l * DD, beta1 + l * DD, n1F, n1B);

        gemm_bf16_kernel<<<dim3(MROWS / BM, DD / BN), 256, 0, stream>>>(
            n1B, WfcB + (size_t)l * DD * DD, tF, nullptr, bfc + l * DD, MROWS, DD, DD, 0);

        ln_kernel<<<MROWS / 8, 256, 0, stream>>>(
            tF, n1F, g2 + l * DD, beta2 + l * DD, hF, hB);
    }
}
